// GNNStack_35639638622535
// MI455X (gfx1250) — matrix operand order repaired and checked by reading, not compiled, not run on hardware
//
#include <hip/hip_runtime.h>
#include <hip/hip_bf16.h>
#include <math.h>

#define NN 100000
#define NE 1600000
#define NG 64
#define FH 128

typedef __attribute__((ext_vector_type(16))) __bf16        v16bf;
typedef __attribute__((ext_vector_type(8)))  float         v8f;
typedef __attribute__((ext_vector_type(4)))  unsigned int  u32x4;

union V16U { u32x4 u[2]; v16bf v; };

__device__ __forceinline__ unsigned short f2bf(float f) {
  unsigned int u = __float_as_uint(f);
  u += 0x7FFFu + ((u >> 16) & 1u);          // round-to-nearest-even
  return (unsigned short)(u >> 16);
}

// ---------------- degree / norm ----------------
__global__ void deg_count(const int* __restrict__ dst, float* __restrict__ deg) {
  int e = blockIdx.x * blockDim.x + threadIdx.x;
  if (e < NE) atomicAdd(&deg[dst[e]], 1.0f);
}

__global__ void deg_rsqrt(const float* __restrict__ deg, float* __restrict__ dinv) {
  int i = blockIdx.x * blockDim.x + threadIdx.x;
  if (i < NN) dinv[i] = rsqrtf(deg[i] + 1.0f);  // +1 self-loop
}

// ---------------- fp32 -> bf16 ----------------
__global__ void f32_to_bf16(const float* __restrict__ in,
                            unsigned short* __restrict__ out, int n) {
  int g = blockIdx.x * blockDim.x + threadIdx.x;
  if (g < n) out[g] = f2bf(in[g]);
}

// ---------------- weight pack into WMMA B-fragment order ----------------
// pack[((ntile*4 + kblock)*32 + lane)*16 + j] = bf16(W[kblock*32 + (j>>3)*16 + (lane>>4)*8 + (j&7)][ntile*16 + lane&15])
__global__ void pack_weights(const float* __restrict__ W,
                             unsigned short* __restrict__ pack) {
  int t = threadIdx.x;              // 1024 threads
  int nt   = t >> 7;                // 0..7  column tile
  int kb   = (t >> 5) & 3;          // 0..3  K block of 32
  int lane = t & 31;
  int n    = nt * 16 + (lane & 15);
  int hlf  = lane >> 4;
  unsigned short* o = pack + ((size_t)((nt * 4 + kb) * 32 + lane)) * 16;
#pragma unroll
  for (int j = 0; j < 16; ++j) {
    int k = kb * 32 + (j >> 3) * 16 + hlf * 8 + (j & 7);
    o[j] = f2bf(W[k * FH + n]);
  }
}

// ---------------- WMMA GEMM: hl[N,128] = hb[N,128] @ W[128,128] ----------------
__global__ void __launch_bounds__(128) gcn_gemm_bf16(
    const unsigned short* __restrict__ hb,
    const unsigned short* __restrict__ bpack,
    float* __restrict__ hl) {
  int wave = blockIdx.x * 4 + (threadIdx.x >> 5);   // one 16-row tile per wave
  if (wave >= NN / 16) return;                      // wave-uniform guard, EXEC stays full
  int lane = threadIdx.x & 31;
  int m    = lane & 15;
  int hlf  = lane >> 4;
  const unsigned short* arow = hb + (size_t)(wave * 16 + m) * FH;

  v8f acc[8];
  v8f zero = {0.f, 0.f, 0.f, 0.f, 0.f, 0.f, 0.f, 0.f};
#pragma unroll
  for (int t = 0; t < 8; ++t) acc[t] = zero;

#pragma unroll
  for (int kb = 0; kb < 4; ++kb) {
    V16U a;   // A 16x32 bf16: lane holds K = kb*32 + hlf*8 + [0..7] and +16
    a.u[0] = *(const u32x4*)(arow + kb * 32 + hlf * 8);
    a.u[1] = *(const u32x4*)(arow + kb * 32 + 16 + hlf * 8);
#pragma unroll
    for (int t = 0; t < 8; ++t) {
      const unsigned short* bp = bpack + ((size_t)(t * 4 + kb) * 32 + lane) * 16;
      V16U b;
      b.u[0] = *(const u32x4*)(bp);
      b.u[1] = *(const u32x4*)(bp + 8);
      acc[t] = __builtin_amdgcn_wmma_f32_16x16x32_bf16(
          false, a.v, false, b.v, (short)0, acc[t], false, false);
    }
  }

  int row_base = wave * 16 + hlf * 8;   // C/D layout: lane -> col m, rows hlf*8 + i
#pragma unroll
  for (int t = 0; t < 8; ++t) {
#pragma unroll
    for (int i = 0; i < 8; ++i) {
      hl[(size_t)(row_base + i) * FH + t * 16 + m] = acc[t][i];
    }
  }
}

// ---------------- edge gather/scale/scatter-add ----------------
__global__ void __launch_bounds__(256) edge_scatter(
    const int* __restrict__ src, const int* __restrict__ dst,
    const float* __restrict__ dinv, const float* __restrict__ hl,
    float* __restrict__ agg) {
  long long g = (long long)blockIdx.x * blockDim.x + threadIdx.x;
  int e = (int)(g >> 5);
  if (e >= NE) return;
  int part = (int)(g & 31);
  int s = src[e], d = dst[e];
  float nm = dinv[s] * dinv[d];
  float4 v = *((const float4*)(hl + (size_t)s * FH) + part);
  float* ap = agg + (size_t)d * FH + part * 4;
  atomicAdd(ap + 0, v.x * nm);
  atomicAdd(ap + 1, v.y * nm);
  atomicAdd(ap + 2, v.z * nm);
  atomicAdd(ap + 3, v.w * nm);
}

// ---------------- epilogue: self-loop + bias + relu (+ bf16 for next layer) ----------------
__global__ void bias_relu_cvt(float* __restrict__ agg, const float* __restrict__ hl,
                              const float* __restrict__ dinv,
                              const float* __restrict__ bias,
                              unsigned short* __restrict__ hb, int n) {
  int g = blockIdx.x * blockDim.x + threadIdx.x;
  if (g >= n) return;
  int node = g >> 7, f = g & 127;
  float di = dinv[node];
  float v = agg[g] + hl[g] * di * di + bias[f];
  v = fmaxf(v, 0.0f);
  agg[g] = v;          // in-place fp32 activation
  hb[g] = f2bf(v);     // bf16 feed for next GEMM
}

// ---------------- segment max pooling (values >= 0 -> int bit compare valid) ----
__global__ void pool_max(const float* __restrict__ h, const int* __restrict__ batch,
                         int* __restrict__ pooled_bits) {
  long long g = (long long)blockIdx.x * blockDim.x + threadIdx.x;
  int node = (int)(g >> 5);
  if (node >= NN) return;
  int part = (int)(g & 31);
  int gr = batch[node];
  float4 v = *((const float4*)(h + (size_t)node * FH) + part);
  int* pp = pooled_bits + gr * FH + part * 4;
  atomicMax(pp + 0, __float_as_int(v.x));
  atomicMax(pp + 1, __float_as_int(v.y));
  atomicMax(pp + 2, __float_as_int(v.z));
  atomicMax(pp + 3, __float_as_int(v.w));
}

// ---------------- MLP head + log_softmax ----------------
__global__ void __launch_bounds__(128) head_kernel(
    const float* __restrict__ pooled,
    const float* __restrict__ Wp1, const float* __restrict__ bp1,
    const float* __restrict__ Wp2, const float* __restrict__ bp2,
    float* __restrict__ out) {
  __shared__ float t1[FH];
  __shared__ float lg[2];
  int g = blockIdx.x;
  int j = threadIdx.x;
  float s = bp1[j];
  for (int k = 0; k < FH; ++k) s += pooled[g * FH + k] * Wp1[k * FH + j];
  t1[j] = s;
  __syncthreads();
  if (j < 2) {
    float s2 = bp2[j];
    for (int k = 0; k < FH; ++k) s2 += t1[k] * Wp2[k * 2 + j];
    lg[j] = s2;
  }
  __syncthreads();
  if (j == 0) {
    float l0 = lg[0], l1 = lg[1];
    float mx = fmaxf(l0, l1);
    float lse = mx + logf(expf(l0 - mx) + expf(l1 - mx));
    out[g * 2 + 0] = l0 - lse;
    out[g * 2 + 1] = l1 - lse;
  }
}

extern "C" void kernel_launch(void* const* d_in, const int* in_sizes, int n_in,
                              void* d_out, int out_size, void* d_ws, size_t ws_size,
                              hipStream_t stream) {
  (void)in_sizes; (void)n_in; (void)out_size; (void)ws_size;
  const float* x    = (const float*)d_in[0];
  const int*   ei   = (const int*)d_in[1];
  const int*   bat  = (const int*)d_in[2];
  const float* W[3] = {(const float*)d_in[3], (const float*)d_in[5], (const float*)d_in[7]};
  const float* B[3] = {(const float*)d_in[4], (const float*)d_in[6], (const float*)d_in[8]};
  const float* Wp1  = (const float*)d_in[9];
  const float* bp1  = (const float*)d_in[10];
  const float* Wp2  = (const float*)d_in[11];
  const float* bp2  = (const float*)d_in[12];
  float* out = (float*)d_out;

  // workspace carve-out (256B aligned): ~129 MB total
  char* w = (char*)d_ws;
  size_t off = 0;
  auto alloc = [&](size_t bytes) -> void* {
    void* p = w + off;
    off = (off + bytes + 255) & ~(size_t)255;
    return p;
  };
  float*          hl     = (float*)alloc((size_t)NN * FH * 4);
  float*          agg    = (float*)alloc((size_t)NN * FH * 4);
  unsigned short* hb     = (unsigned short*)alloc((size_t)NN * FH * 2);
  float*          deg    = (float*)alloc((size_t)NN * 4);
  float*          dinv   = (float*)alloc((size_t)NN * 4);
  unsigned short* pack   = (unsigned short*)alloc((size_t)8 * 4 * 32 * 16 * 2);
  float*          pooled = (float*)alloc((size_t)NG * FH * 4);

  const int* src = ei;
  const int* dst = ei + NE;

  // degrees & symmetric norm
  hipMemsetAsync(deg, 0, (size_t)NN * 4, stream);
  deg_count<<<NE / 256, 256, 0, stream>>>(dst, deg);
  deg_rsqrt<<<(NN + 255) / 256, 256, 0, stream>>>(deg, dinv);

  // initial activations -> bf16
  f32_to_bf16<<<NN * FH / 256, 256, 0, stream>>>(x, hb, NN * FH);

  for (int l = 0; l < 3; ++l) {
    pack_weights<<<1, 1024, 0, stream>>>(W[l], pack);
    gcn_gemm_bf16<<<(NN / 16 + 3) / 4, 128, 0, stream>>>(hb, pack, hl);
    hipMemsetAsync(agg, 0, (size_t)NN * FH * 4, stream);
    edge_scatter<<<(int)(((long long)NE * 32) / 256), 256, 0, stream>>>(src, dst, dinv, hl, agg);
    bias_relu_cvt<<<NN * FH / 256, 256, 0, stream>>>(agg, hl, dinv, B[l], hb, NN * FH);
  }

  hipMemsetAsync(pooled, 0, (size_t)NG * FH * 4, stream);
  pool_max<<<(int)(((long long)NN * 32 + 255) / 256), 256, 0, stream>>>(agg, bat, (int*)pooled);
  head_kernel<<<NG, 128, 0, stream>>>(pooled, Wp1, bp1, Wp2, bp2, out);
}
